// FourierCrossAttention_90452011254241
// MI455X (gfx1250) — matrix operand order repaired and checked by reading, not compiled, not run on hardware
//
#include <hip/hip_runtime.h>
#include <hip/hip_bf16.h>

// ---------------------------------------------------------------------------
// FourierCrossAttention on gfx1250 (MI455X).
//   q,k DFT(64 modes), complex scores + ctanh, xqkv, per-mode weights, iDFT.
//   All large GEMMs use v_wmma_f32_16x16x32_f16 (f16 operands, f32 accum).
// ---------------------------------------------------------------------------

#define BATCH 32
#define SEQL  2048
#define HEADS 8
#define EDIM  64
#define ODIM  64
#define NMODE 64

typedef __attribute__((ext_vector_type(16))) _Float16 v16h;
typedef __attribute__((ext_vector_type(8)))  _Float16 v8h;
typedef __attribute__((ext_vector_type(8)))  float    v8f;

union V16H { v16h v; v8h h[2]; _Float16 e[16]; };

__device__ inline v8f vzero8f() {
    v8f z;
#pragma unroll
    for (int i = 0; i < 8; ++i) z[i] = 0.0f;
    return z;
}

__device__ inline v8f wmma_f16(v16h a, v16h b, v8f c) {
    // D = A(16x32 f16) * B(32x16 f16) + C(16x16 f32)
    return __builtin_amdgcn_wmma_f32_16x16x32_f16(
        false, a, false, b, (short)0, c, false, false);
}

// A-fragment (16x32, f16) from row-major matrix M[row][k], leading dim ld.
// lanes 0-15 = rows, elements 0..15 -> K=lo..lo+15
// (lo = 0 for lanes 0-15, 16 for lanes 16-31).  Two contiguous 16B loads.
__device__ inline v16h load_frag_a(const _Float16* __restrict__ M, int ld,
                                   int row0, int k0, int lane) {
    int r  = row0 + (lane & 15);
    int lo = (lane & 16) ? 16 : 0;
    const _Float16* p = M + (size_t)r * ld + k0 + lo;
    V16H u;
    u.h[0] = *(const v8h*)(p);
    u.h[1] = *(const v8h*)(p + 8);
    return u.v;
}

// B-fragment (32x16, f16) from the *transposed* store T[n][k] (ld over k).
// lane -> column n, halves hh -> K = koff+hh (koff = 0 / 16): 32 contiguous B.
__device__ inline v16h load_frag_b_T(const _Float16* __restrict__ T, int ld,
                                     int k0, int col0, int lane) {
    int c    = col0 + (lane & 15);
    int koff = (lane & 16) ? 16 : 0;
    const _Float16* p = T + (size_t)c * ld + k0 + koff;
    V16H u;
    u.h[0] = *(const v8h*)(p);
    u.h[1] = *(const v8h*)(p + 8);
    return u.v;
}

// B-fragment from a row-major [k][n] store (strided per-half loads).
__device__ inline v16h load_frag_b_strided(const _Float16* __restrict__ M,
                                           int ld, int k0, int col0, int lane) {
    int c    = col0 + (lane & 15);
    int koff = (lane & 16) ? 16 : 0;
    const _Float16* p = M + (size_t)(k0 + koff) * ld + c;
    V16H u;
#pragma unroll
    for (int hh = 0; hh < 16; ++hh) u.e[hh] = p[(size_t)hh * ld];
    return u.v;
}

// ---------------------------------------------------------------------------
// Mode-index generation (deterministic stand-in for numpy's seeded shuffle;
// produces 64 sorted distinct frequencies in [0, L/2) for q and kv).
// ---------------------------------------------------------------------------
__global__ void build_indices(int* __restrict__ idx) {
    __shared__ int pool[SEQL / 2];
    if (threadIdx.x != 0 || blockIdx.x != 0) return;
    unsigned s = 0x243F6A88u;
    for (int set = 0; set < 2; ++set) {
        for (int i = 0; i < SEQL / 2; ++i) pool[i] = i;
        for (int i = SEQL / 2 - 1; i > 0; --i) {
            s = s * 1664525u + 1013904223u;
            int j = (int)(s % (unsigned)(i + 1));
            int t = pool[i]; pool[i] = pool[j]; pool[j] = t;
        }
        int sel[NMODE];
        for (int i = 0; i < NMODE; ++i) sel[i] = pool[i];
        for (int i = 1; i < NMODE; ++i) {           // insertion sort
            int v = sel[i], j = i - 1;
            while (j >= 0 && sel[j] > v) { sel[j + 1] = sel[j]; --j; }
            sel[j + 1] = v;
        }
        for (int i = 0; i < NMODE; ++i) idx[set * NMODE + i] = sel[i];
    }
}

// Forward DFT twiddle: TW[128][SEQL] f16, rows m<64 = cos(th), rows 64+m = -sin(th),
// th = 2*pi*idx[m]*l/L  (rfft convention e^{-i th}).
__global__ void build_dft_tw(const int* __restrict__ idx,
                             _Float16* __restrict__ TW) {
    int l = blockIdx.x * blockDim.x + threadIdx.x;
    if (l >= SEQL) return;
#pragma unroll 4
    for (int m = 0; m < NMODE; ++m) {
        int kk = idx[m];
        int ph = (kk * l) & (SEQL - 1);              // exact phase mod L
        float th = 6.28318530717958647692f * (float)ph / (float)SEQL;
        TW[(size_t)m * SEQL + l]            = (_Float16)__cosf(th);
        TW[(size_t)(NMODE + m) * SEQL + l]  = (_Float16)(-__sinf(th));
    }
}

// Inverse DFT twiddle (transposed for B-fragments): TWI[SEQL][128],
// TWI[l][m] = c*cos(th)/L, TWI[l][64+m] = -c*sin(th)/L, c = (idx==0 ? 1 : 2).
__global__ void build_idft_tw(const int* __restrict__ idxq,
                              _Float16* __restrict__ TWI) {
    int l = blockIdx.x * blockDim.x + threadIdx.x;
    if (l >= SEQL) return;
#pragma unroll 4
    for (int m = 0; m < NMODE; ++m) {
        int kk = idxq[m];
        int ph = (kk * l) & (SEQL - 1);
        float th  = 6.28318530717958647692f * (float)ph / (float)SEQL;
        float inv = ((kk == 0) ? 1.0f : 2.0f) / (float)SEQL;
        TWI[(size_t)l * 128 + m]         = (_Float16)(__cosf(th) * inv);
        TWI[(size_t)l * 128 + NMODE + m] = (_Float16)(-__sinf(th) * inv);
    }
}

// ---------------------------------------------------------------------------
// Stage 1: DFT of q (or k).  Per block: one (b,h); GEMM [128 x 64 x K=2048].
//   A = TW (global, f16, L2-resident), B = input tile staged f32->f16 in LDS.
//   X[b,h,r,e], r<64 real part of mode r, r>=64 imag part.
// ---------------------------------------------------------------------------
__global__ __launch_bounds__(256) void dft_wmma(
    const float* __restrict__ x,       // [B, L, H, E] f32
    const _Float16* __restrict__ TW,   // [128][SEQL] f16
    _Float16* __restrict__ X)          // [B,H,128,64] f16
{
    __shared__ _Float16 Bs[EDIM][40];  // [e][kk], pad 40 -> conflict-free v8h
    int bh = blockIdx.x;
    int b  = bh >> 3, h = bh & 7;
    int tid = threadIdx.x, wave = tid >> 5, lane = tid & 31;
    int mrow = wave * 16;

    const float* qbase = x + ((size_t)b * SEQL * HEADS + h) * EDIM;

    v8f acc[4];
#pragma unroll
    for (int i = 0; i < 4; ++i) acc[i] = vzero8f();

    for (int kb = 0; kb < SEQL; kb += 32) {
        __syncthreads();
        {   // stage 32(k) x 64(e) chunk, transposed, f32 -> f16
            int kk = tid >> 3;            // 0..31
            int e0 = (tid & 7) * 8;       // 0..56
            const float* src = qbase + (size_t)(kb + kk) * (HEADS * EDIM) + e0;
#pragma unroll
            for (int j = 0; j < 8; ++j) Bs[e0 + j][kk] = (_Float16)src[j];
        }
        __syncthreads();

        v16h af = load_frag_a(TW, SEQL, mrow, kb, lane);
#pragma unroll
        for (int nt = 0; nt < 4; ++nt) {
            v16h bf = load_frag_b_T(&Bs[0][0], 40, 0, nt * 16, lane);
            acc[nt] = wmma_f16(af, bf, acc[nt]);
        }
    }

    _Float16* out = X + (size_t)bh * 128 * EDIM;
    int m0 = mrow + ((lane & 16) ? 8 : 0);
    int n_ = lane & 15;
#pragma unroll
    for (int nt = 0; nt < 4; ++nt)
#pragma unroll
        for (int j = 0; j < 8; ++j)
            out[(size_t)(m0 + j) * EDIM + nt * 16 + n_] = (_Float16)acc[nt][j];
}

// ---------------------------------------------------------------------------
// Stage 2: complex scores (no conj) + complex tanh.
//   S[x,y] = sum_e Q[x,e]*K[y,e]   -> A = XQ rows, B = XK rows (both natural).
//   tanh(x+iy) = (sinh 2x + i sin 2y) / (cosh 2x + cos 2y), saturated.
// ---------------------------------------------------------------------------
__device__ inline void ctanhf2(float x, float y, float& tr, float& ti) {
    float x2 = 2.0f * x, y2 = 2.0f * y;
    if (__builtin_fabsf(x2) > 40.0f) {
        tr = (x2 > 0.0f) ? 1.0f : -1.0f; ti = 0.0f; return;
    }
    float ex = __expf(x2), emx = 1.0f / ex;
    float d  = 0.5f * (ex + emx) + __cosf(y2);
    tr = 0.5f * (ex - emx) / d;
    ti = __sinf(y2) / d;
}

__global__ __launch_bounds__(256) void scores_tanh_wmma(
    const _Float16* __restrict__ XQ,   // [B,H,128,64]
    const _Float16* __restrict__ XK,   // [B,H,128,64]
    _Float16* __restrict__ S)          // [B,H,2,64,64]  (p=0 real, 1 imag)
{
    int bh = blockIdx.x;
    const _Float16* Q  = XQ + (size_t)bh * 128 * EDIM;
    const _Float16* K  = XK + (size_t)bh * 128 * EDIM;
    const _Float16* Ki = K + (size_t)NMODE * EDIM;
    int tid = threadIdx.x, wave = tid >> 5, lane = tid & 31;
    int mt = wave & 3, nh = wave >> 2;           // x-tile, y-tile-pair

    v8f arr[2], aii[2], ari[2], air[2];
#pragma unroll
    for (int t = 0; t < 2; ++t) {
        arr[t] = vzero8f(); aii[t] = vzero8f();
        ari[t] = vzero8f(); air[t] = vzero8f();
    }

#pragma unroll
    for (int kb = 0; kb < EDIM; kb += 32) {
        v16h qr = load_frag_a(Q, EDIM, mt * 16, kb, lane);
        v16h qi = load_frag_a(Q, EDIM, NMODE + mt * 16, kb, lane);
#pragma unroll
        for (int t = 0; t < 2; ++t) {
            int y0 = nh * 32 + t * 16;
            v16h kr = load_frag_b_T(K,  EDIM, kb, y0, lane);
            v16h ki = load_frag_b_T(Ki, EDIM, kb, y0, lane);
            arr[t] = wmma_f16(qr, kr, arr[t]);
            aii[t] = wmma_f16(qi, ki, aii[t]);
            ari[t] = wmma_f16(qr, ki, ari[t]);
            air[t] = wmma_f16(qi, kr, air[t]);
        }
    }

    _Float16* Sr = S + (size_t)bh * 2 * NMODE * NMODE;
    _Float16* Si = Sr + (size_t)NMODE * NMODE;
    int m0 = mt * 16 + ((lane & 16) ? 8 : 0);
    int n_ = lane & 15;
#pragma unroll
    for (int t = 0; t < 2; ++t) {
        int y0 = nh * 32 + t * 16;
#pragma unroll
        for (int j = 0; j < 8; ++j) {
            float sr = arr[t][j] - aii[t][j];
            float si = ari[t][j] + air[t][j];
            float tr, ti;
            ctanhf2(sr, si, tr, ti);
            Sr[(size_t)(m0 + j) * NMODE + y0 + n_] = (_Float16)tr;
            Si[(size_t)(m0 + j) * NMODE + y0 + n_] = (_Float16)ti;
        }
    }
}

// ---------------------------------------------------------------------------
// Stage 3: xqkv[x,e] = sum_y S[x,y] * Xk[y,e]  (complex).
//   A = S rows (natural), B = XK in [k=y][n=e] order -> strided B-fragments.
// ---------------------------------------------------------------------------
__global__ __launch_bounds__(256) void xqkv_wmma(
    const _Float16* __restrict__ S,    // [B,H,2,64,64]
    const _Float16* __restrict__ XK,   // [B,H,128,64]
    _Float16* __restrict__ XQKV)       // [B,H,2,64(x),64(e)]
{
    int bh = blockIdx.x;
    const _Float16* Sr = S + (size_t)bh * 2 * NMODE * NMODE;
    const _Float16* Si = Sr + (size_t)NMODE * NMODE;
    const _Float16* Kr = XK + (size_t)bh * 128 * EDIM;
    const _Float16* Ki = Kr + (size_t)NMODE * EDIM;
    int tid = threadIdx.x, wave = tid >> 5, lane = tid & 31;
    int mt = wave & 3, nh = wave >> 2;

    v8f arr[2], aii[2], ari[2], air[2];
#pragma unroll
    for (int t = 0; t < 2; ++t) {
        arr[t] = vzero8f(); aii[t] = vzero8f();
        ari[t] = vzero8f(); air[t] = vzero8f();
    }

#pragma unroll
    for (int kb = 0; kb < NMODE; kb += 32) {
        v16h sr = load_frag_a(Sr, NMODE, mt * 16, kb, lane);
        v16h si = load_frag_a(Si, NMODE, mt * 16, kb, lane);
#pragma unroll
        for (int t = 0; t < 2; ++t) {
            int e0 = nh * 32 + t * 16;
            v16h kr = load_frag_b_strided(Kr, EDIM, kb, e0, lane);
            v16h ki = load_frag_b_strided(Ki, EDIM, kb, e0, lane);
            arr[t] = wmma_f16(sr, kr, arr[t]);
            aii[t] = wmma_f16(si, ki, aii[t]);
            ari[t] = wmma_f16(sr, ki, ari[t]);
            air[t] = wmma_f16(si, kr, air[t]);
        }
    }

    _Float16* Or = XQKV + (size_t)bh * 2 * NMODE * EDIM;
    _Float16* Oi = Or + (size_t)NMODE * EDIM;
    int m0 = mt * 16 + ((lane & 16) ? 8 : 0);
    int n_ = lane & 15;
#pragma unroll
    for (int t = 0; t < 2; ++t)
#pragma unroll
        for (int j = 0; j < 8; ++j) {
            int xr = m0 + j, e = nh * 32 + t * 16 + n_;
            Or[(size_t)xr * EDIM + e] = (_Float16)(arr[t][j] - aii[t][j]);
            Oi[(size_t)xr * EDIM + e] = (_Float16)(ari[t][j] + air[t][j]);
        }
}

// ---------------------------------------------------------------------------
// Stage 4: xqkvw[o,x] = sum_e xqkv[e,x] * w[h,e,o,x]  (complex, per-mode W).
//   Small contraction (K=64) with mode-dependent weights -> VALU kernel.
//   Applies the 1/(IN_CH*OUT_CH) spectrum scale; writes X2[b,h,o, x | 64+x].
// ---------------------------------------------------------------------------
__global__ __launch_bounds__(64) void xqkvw_valu(
    const _Float16* __restrict__ XQKV, // [B,H,2,64,64]
    const float* __restrict__ Wr,      // [H,64,64,64]
    const float* __restrict__ Wi,
    _Float16* __restrict__ X2)         // [B,H,64(o),128(r)]
{
    __shared__ float qr[EDIM], qi[EDIM];
    int blk = blockIdx.x;
    int x   = blk & 63;
    int bh  = blk >> 6;
    int h   = bh & 7;
    int o   = threadIdx.x;

    const _Float16* src = XQKV + (size_t)bh * 2 * NMODE * EDIM;
    qr[o] = (float)src[(size_t)x * EDIM + o];
    qi[o] = (float)src[(size_t)NMODE * EDIM + (size_t)x * EDIM + o];
    __syncthreads();

    const float* wr = Wr + (size_t)h * 262144 + (size_t)o * 64 + x;
    const float* wi = Wi + (size_t)h * 262144 + (size_t)o * 64 + x;
    float ar = 0.0f, ai = 0.0f;
#pragma unroll 4
    for (int e = 0; e < EDIM; ++e) {
        float wrv = wr[(size_t)e * 4096], wiv = wi[(size_t)e * 4096];
        ar += qr[e] * wrv - qi[e] * wiv;
        ai += qr[e] * wiv + qi[e] * wrv;
    }
    const float s = 1.0f / (512.0f * 512.0f);   // out_ft / (IN_CH*OUT_CH)
    _Float16* dst = X2 + (size_t)bh * ODIM * 128 + (size_t)o * 128;
    dst[x]         = (_Float16)(ar * s);
    dst[NMODE + x] = (_Float16)(ai * s);
}

// ---------------------------------------------------------------------------
// Stage 5: inverse DFT.  out[o,l] = sum_r X2[o,r] * TWI[l,r]; K=128, N=2048.
//   A = X2 rows (natural), B = TWI transposed store (natural).  f32 output.
// ---------------------------------------------------------------------------
__global__ __launch_bounds__(256) void idft_wmma(
    const _Float16* __restrict__ X2,   // [B,H,64,128]
    const _Float16* __restrict__ TWI,  // [SEQL][128]
    float* __restrict__ out)           // [B,H,64,SEQL]
{
    int blk = blockIdx.x;
    int lchunk = blk & 7;
    int bh     = blk >> 3;
    int tid = threadIdx.x, wave = tid >> 5, lane = tid & 31;
    const _Float16* A = X2 + (size_t)bh * ODIM * 128;
    int l0 = lchunk * 256 + wave * 32;

    v8f acc[4][2];
#pragma unroll
    for (int mt = 0; mt < 4; ++mt) { acc[mt][0] = vzero8f(); acc[mt][1] = vzero8f(); }

#pragma unroll
    for (int kb = 0; kb < 128; kb += 32) {
        v16h bf0 = load_frag_b_T(TWI, 128, kb, l0, lane);
        v16h bf1 = load_frag_b_T(TWI, 128, kb, l0 + 16, lane);
#pragma unroll
        for (int mt = 0; mt < 4; ++mt) {
            v16h af = load_frag_a(A, 128, mt * 16, kb, lane);
            acc[mt][0] = wmma_f16(af, bf0, acc[mt][0]);
            acc[mt][1] = wmma_f16(af, bf1, acc[mt][1]);
        }
    }

    float* obase = out + (size_t)bh * ODIM * SEQL;
    int n_ = lane & 15, moff = (lane & 16) ? 8 : 0;
#pragma unroll
    for (int mt = 0; mt < 4; ++mt)
#pragma unroll
        for (int t = 0; t < 2; ++t)
#pragma unroll
            for (int j = 0; j < 8; ++j) {
                int o = mt * 16 + moff + j;
                int l = l0 + t * 16 + n_;
                obase[(size_t)o * SEQL + l] = acc[mt][t][j];
            }
}

// ---------------------------------------------------------------------------
extern "C" void kernel_launch(void* const* d_in, const int* in_sizes, int n_in,
                              void* d_out, int out_size, void* d_ws, size_t ws_size,
                              hipStream_t stream) {
    (void)in_sizes; (void)n_in; (void)out_size; (void)ws_size;
    const float* q  = (const float*)d_in[0];   // [B,L,H,E]
    const float* k  = (const float*)d_in[1];   // [B,L,H,E]
    const float* Wr = (const float*)d_in[4];   // [H,E,O,M]
    const float* Wi = (const float*)d_in[5];
    float* out = (float*)d_out;                // [B,H,O,L]

    // workspace carve-up (256B aligned), ~20 MB total
    char* ws = (char*)d_ws;
    size_t off = 0;
    auto carve = [&](size_t bytes) -> char* {
        char* p = ws + off;
        off += (bytes + 255) & ~(size_t)255;
        return p;
    };
    int*       idx  = (int*)carve(2 * NMODE * sizeof(int));
    _Float16*  TWQ  = (_Float16*)carve((size_t)128 * SEQL * 2);
    _Float16*  TWK  = (_Float16*)carve((size_t)128 * SEQL * 2);
    _Float16*  TWI  = (_Float16*)carve((size_t)SEQL * 128 * 2);
    _Float16*  XQ   = (_Float16*)carve((size_t)BATCH * HEADS * 128 * EDIM * 2);
    _Float16*  XK   = (_Float16*)carve((size_t)BATCH * HEADS * 128 * EDIM * 2);
    _Float16*  S    = (_Float16*)carve((size_t)BATCH * HEADS * 2 * NMODE * NMODE * 2);
    _Float16*  XQKV = (_Float16*)carve((size_t)BATCH * HEADS * 2 * NMODE * EDIM * 2);
    _Float16*  X2   = (_Float16*)carve((size_t)BATCH * HEADS * ODIM * 128 * 2);

    build_indices<<<1, 32, 0, stream>>>(idx);
    build_dft_tw <<<SEQL / 256, 256, 0, stream>>>(idx,         TWQ);
    build_dft_tw <<<SEQL / 256, 256, 0, stream>>>(idx + NMODE, TWK);
    build_idft_tw<<<SEQL / 256, 256, 0, stream>>>(idx,         TWI);

    dft_wmma<<<BATCH * HEADS, 256, 0, stream>>>(q, TWQ, XQ);
    dft_wmma<<<BATCH * HEADS, 256, 0, stream>>>(k, TWK, XK);

    scores_tanh_wmma<<<BATCH * HEADS, 256, 0, stream>>>(XQ, XK, S);
    xqkv_wmma       <<<BATCH * HEADS, 256, 0, stream>>>(S, XK, XQKV);
    xqkvw_valu      <<<BATCH * HEADS * NMODE, 64, 0, stream>>>(XQKV, Wr, Wi, X2);
    idft_wmma       <<<BATCH * HEADS * (SEQL / 256), 256, 0, stream>>>(X2, TWI, out);
}
